// dgmslice_ps2_51668456571071
// MI455X (gfx1250) — matrix operand order repaired and checked by reading, not compiled, not run on hardware
//
#include <hip/hip_runtime.h>
#include <hip/hip_bf16.h>

typedef _Float16 f16;
typedef __attribute__((ext_vector_type(16))) _Float16 v16h;
typedef __attribute__((ext_vector_type(8)))  _Float16 v8h;
typedef __attribute__((ext_vector_type(8)))  float    v8f;

#define NL    64      // L
#define NEIGV 128     // NEIG
#define NTOT  8192    // L*NEIG
#define NSIMP 16384
#define NF    8       // F
#define HUNI  512     // HU
#define NHL   3       // HL
#define EPSB  1e-5f

// ---------------------------------------------------------------------------
// Generic column mean / biased-variance over `rows` for `cols` columns,
// batched over blockIdx.y vectors.  x layout: [v][rows][cols].
// ---------------------------------------------------------------------------
__global__ __launch_bounds__(256) void k_colstats(const float* __restrict__ x,
                                                  float* __restrict__ mu,
                                                  float* __restrict__ var,
                                                  int rows, int cols) {
  int col = blockIdx.x;
  int v   = blockIdx.y;
  const float* p = x + (size_t)v * rows * cols + col;
  float s = 0.f, q = 0.f;
  for (int i = threadIdx.x; i < rows; i += 256) {
    float t = p[(size_t)i * cols];
    s += t; q += t * t;
  }
  __shared__ float rs[256], rq[256];
  rs[threadIdx.x] = s; rq[threadIdx.x] = q;
  __syncthreads();
  for (int o = 128; o > 0; o >>= 1) {
    if (threadIdx.x < o) { rs[threadIdx.x] += rs[threadIdx.x + o];
                           rq[threadIdx.x] += rq[threadIdx.x + o]; }
    __syncthreads();
  }
  if (threadIdx.x == 0) {
    float m = rs[0] / (float)rows;
    mu [(size_t)v * cols + col] = m;
    var[(size_t)v * cols + col] = rq[0] / (float)rows - m * m;
  }
}

// ---------------------------------------------------------------------------
// Input layer: BN of (e*W+b) collapses to W*(e-mu_e)*rsqrt(W^2*var_e+eps).
// act[f][i][h]  (f16)
// ---------------------------------------------------------------------------
__global__ void k_inlayer(const float* __restrict__ e,
                          const float* __restrict__ emu,
                          const float* __restrict__ evar,
                          const float* __restrict__ Win,
                          const float* __restrict__ gin,
                          const float* __restrict__ betain,
                          f16* __restrict__ act) {
  size_t total = (size_t)NF * NTOT * HUNI;
  float mu = emu[0], var = evar[0];
  for (size_t idx = (size_t)blockIdx.x * blockDim.x + threadIdx.x; idx < total;
       idx += (size_t)gridDim.x * blockDim.x) {
    int h = (int)(idx & (HUNI - 1));
    size_t t = idx >> 9;
    int i = (int)(t & (NTOT - 1));
    int f = (int)(t >> 13);
    float W = Win[f * HUNI + h];
    float x = W * (e[i] - mu) * rsqrtf(W * W * var + EPSB);
    float a = gin[f * HUNI + h] * x + betain[f * HUNI + h];
    act[idx] = (f16)fmaxf(a, 0.f);
  }
}

// ---------------------------------------------------------------------------
// Hidden-layer GEMM: Y[f] = act[f] (8192x512, f16) @ W[f][layer]^T + bias.
// 64x32 workgroup tile, 8 waves, each wave one 16x16 WMMA accumulator,
// K looped 512 in steps of 32 via v_wmma_f32_16x16x32_f16.
//
// Double-buffered LDS pipeline: tile t+1 is fetched with the CDNA5 async
// global->LDS DMA (ASYNCcnt) + a plain global B load while tile t's WMMA
// runs; s_wait_asynccnt is deferred to just before the single barrier, so
// the copy latency hides under compute.
// ---------------------------------------------------------------------------
__global__ __launch_bounds__(256) void k_gemm512(const f16* __restrict__ act,
                                                 const float* __restrict__ Wh,
                                                 const float* __restrict__ bh,
                                                 float* __restrict__ Y,
                                                 int layer) {
  __shared__ f16 As[2][64][32];   // [buf][row i][k]
  __shared__ f16 Bs[2][32][32];   // [buf][row n][k]  (== W row-major slice)
  const int f    = blockIdx.z;
  const int i0   = blockIdx.x * 64;
  const int n0   = blockIdx.y * 32;
  const int tid  = threadIdx.x;
  const int lane = tid & 31;
  const int wid  = tid >> 5;
  const int wm   = wid & 3;    // 4 waves along M
  const int wn   = wid >> 2;   // 2 waves along N
  const f16*   A  = act + (size_t)f * NTOT * HUNI;
  const float* Wf = Wh + (((size_t)f * NHL + layer) * HUNI) * HUNI;

  const int arowld = tid >> 2;            // A-tile loader row
  const int acolld = (tid & 3) * 8;       // 8 halfs (16B) per thread
  const int nrowld = tid >> 3;            // B-tile loader row (n)
  const int kcolld = (tid & 7) * 4;       // 4 f32 per thread

  // per-lane LDS byte addresses of this thread's A-tile slots (both buffers)
  // (flat shared pointer: addr[31:0] is the LDS offset)
  const unsigned a_lds0 = (unsigned)(unsigned long long)(const void*)&As[0][arowld][acolld];
  const unsigned a_lds1 = (unsigned)(unsigned long long)(const void*)&As[1][arowld][acolld];

  // ---- prologue: stage tile 0 into buffer 0 ----
  {
    const f16* asrc = A + (size_t)(i0 + arowld) * HUNI + acolld;
    asm volatile("global_load_async_to_lds_b128 %0, %1, off"
                 :: "v"(a_lds0), "v"((unsigned long long)asrc) : "memory");
    const float* bsrc = Wf + (size_t)(n0 + nrowld) * HUNI + kcolld;
    float4 bq = *(const float4*)bsrc;
    Bs[0][nrowld][kcolld + 0] = (f16)bq.x;
    Bs[0][nrowld][kcolld + 1] = (f16)bq.y;
    Bs[0][nrowld][kcolld + 2] = (f16)bq.z;
    Bs[0][nrowld][kcolld + 3] = (f16)bq.w;
  }
  asm volatile("s_wait_asynccnt 0x0" ::: "memory");
  __syncthreads();

  v8f c = {};
#pragma unroll
  for (int t = 0; t < HUNI / 32; ++t) {
    const int  cur     = t & 1;
    const int  nxt     = cur ^ 1;
    const bool hasNext = (t + 1) < (HUNI / 32);

    // kick off tile t+1: async A DMA into the idle buffer + B global load
    float4 bq{};
    if (hasNext) {
      const int  kn   = (t + 1) * 32;
      const f16* asrc = A + (size_t)(i0 + arowld) * HUNI + kn + acolld;
      const unsigned ldsn = (cur == 0) ? a_lds1 : a_lds0;
      asm volatile("global_load_async_to_lds_b128 %0, %1, off"
                   :: "v"(ldsn), "v"((unsigned long long)asrc) : "memory");
      bq = *(const float4*)(Wf + (size_t)(n0 + nrowld) * HUNI + kn + kcolld);
      if (t + 2 < HUNI / 32) __builtin_prefetch(asrc + 32, 0, 1);
    }

    // ---- compute on tile t (buffer cur) ----
    // A fragment: lane<16 -> row lane, K {0..7,16..23}; lane>=16 -> K {8..15,24..31}
    const int arow  = wm * 16 + (lane & 15);
    const int abase = (lane < 16) ? 0 : 8;
    v8h alo = *(const v8h*)&As[cur][arow][abase];
    v8h ahi = *(const v8h*)&As[cur][arow][abase + 16];
    // B fragment: lane<16 -> col lane, K {0..7,16..23}; lane>=16 -> K {8..15,24..31}
    const int bcol  = wn * 16 + (lane & 15);
    const int bbase = (lane < 16) ? 0 : 8;
    v8h blo = *(const v8h*)&Bs[cur][bcol][bbase];
    v8h bhi = *(const v8h*)&Bs[cur][bcol][bbase + 16];
    v16h a, b;
    for (int j = 0; j < 8; ++j) {
      a[j] = alo[j]; a[j + 8] = ahi[j];
      b[j] = blo[j]; b[j + 8] = bhi[j];
    }
    c = __builtin_amdgcn_wmma_f32_16x16x32_f16(false, a, false, b,
                                               (short)0, c, false, false);

    // finish staging tile t+1 into the idle buffer, then one barrier
    if (hasNext) {
      Bs[nxt][nrowld][kcolld + 0] = (f16)bq.x;
      Bs[nxt][nrowld][kcolld + 1] = (f16)bq.y;
      Bs[nxt][nrowld][kcolld + 2] = (f16)bq.z;
      Bs[nxt][nrowld][kcolld + 3] = (f16)bq.w;
      asm volatile("s_wait_asynccnt 0x0" ::: "memory");
    }
    __syncthreads();
  }

  // epilogue: C layout VGPR r -> M = r + (lane>=16 ? 8 : 0), N = lane&15
  const int ccol  = n0 + wn * 16 + (lane & 15);
  const float bv  = bh[((size_t)f * NHL + layer) * HUNI + ccol];
  const int rbase = i0 + wm * 16 + ((lane >> 4) ? 8 : 0);
  float* Yf = Y + (size_t)f * NTOT * HUNI;
  for (int r = 0; r < 8; ++r)
    Yf[(size_t)(rbase + r) * HUNI + ccol] = c[r] + bv;
}

// ---------------------------------------------------------------------------
// Hidden BN + ReLU + cast to f16 activations (in place over act buffer).
// ---------------------------------------------------------------------------
__global__ void k_bnrelu(const float* __restrict__ Y,
                         const float* __restrict__ mu,
                         const float* __restrict__ var,
                         const float* __restrict__ gh,
                         const float* __restrict__ betah,
                         f16* __restrict__ act, int layer) {
  size_t total = (size_t)NF * NTOT * HUNI;
  for (size_t idx = (size_t)blockIdx.x * blockDim.x + threadIdx.x; idx < total;
       idx += (size_t)gridDim.x * blockDim.x) {
    int h = (int)(idx & (HUNI - 1));
    int f = (int)(idx >> 22);  // idx / (8192*512)
    float m = mu[f * HUNI + h];
    float v = var[f * HUNI + h];
    size_t po = ((size_t)f * NHL + layer) * HUNI + h;
    float x = gh[po] * (Y[idx] - m) * rsqrtf(v + EPSB) + betah[po];
    act[idx] = (f16)fmaxf(x, 0.f);
  }
}

// ---------------------------------------------------------------------------
// Output layer dot: yout[f][i] = act[f][i][:] . Wout[f] + bout[f].
// One wave per row (wave32 shuffle reduction).
// ---------------------------------------------------------------------------
__global__ __launch_bounds__(256) void k_outdot(const f16* __restrict__ act,
                                                const float* __restrict__ Wout,
                                                const float* __restrict__ bout,
                                                float* __restrict__ yout) {
  int f    = blockIdx.y;
  int lane = threadIdx.x & 31;
  int wid  = threadIdx.x >> 5;
  int row  = blockIdx.x * 8 + wid;
  const f16*   ar = act + ((size_t)f * NTOT + row) * HUNI;
  const float* w  = Wout + f * HUNI;
  float s = 0.f;
  for (int k = lane; k < HUNI; k += 32) s += (float)ar[k] * w[k];
  for (int o = 16; o > 0; o >>= 1) s += __shfl_xor(s, o, 32);
  if (lane == 0) yout[(size_t)f * NTOT + row] = s + bout[f];
}

// ---------------------------------------------------------------------------
// BN+ReLU of output scalar; store as WMMA-ready A fragments trA[l][f][n] f16.
// ---------------------------------------------------------------------------
__global__ void k_trbn(const float* __restrict__ yout,
                       const float* __restrict__ mu,
                       const float* __restrict__ var,
                       const float* __restrict__ gout,
                       const float* __restrict__ betaout,
                       f16* __restrict__ trA) {
  int idx = blockIdx.x * blockDim.x + threadIdx.x;
  if (idx >= NF * NTOT) return;
  int f = idx >> 13, i = idx & (NTOT - 1);
  float x = gout[f] * (yout[idx] - mu[f]) * rsqrtf(var[f] + EPSB) + betaout[f];
  x = fmaxf(x, 0.f);
  int l = i >> 7, n = i & (NEIGV - 1);
  trA[(size_t)l * (NF * NEIGV) + f * NEIGV + n] = (f16)x;
}

// ---------------------------------------------------------------------------
// fs[f][l][s] = sum_n tr[f,l,n] * eigvec_sq[l,n,s]   via WMMA.
// M=8 (padded to 16 with zero rows in the fragment), K=128, N tiled by 16.
// Streams the 537 MB eigvec tensor exactly once (HBM-bound pass).
// ---------------------------------------------------------------------------
__global__ __launch_bounds__(256) void k_fs(const f16* __restrict__ trA,
                                            const float* __restrict__ eig,
                                            float* __restrict__ fs) {
  int l    = blockIdx.y;
  int lane = threadIdx.x & 31;
  int wid  = threadIdx.x >> 5;
  int s0   = blockIdx.x * 128 + wid * 16;
  int fRow = lane & 15;
  const f16*   trl = trA + (size_t)l * (NF * NEIGV);
  const float* el  = eig + (size_t)l * NEIGV * NSIMP;
  v8f c = {};
  for (int k0 = 0; k0 < NEIGV; k0 += 32) {
    v16h a, b;
    int abase = k0 + ((lane < 16) ? 0 : 8);
    if (fRow < NF) {
      const f16* ap = trl + fRow * NEIGV;
      for (int j = 0; j < 8; ++j) { a[j] = ap[abase + j]; a[j + 8] = ap[abase + 16 + j]; }
    } else {
      for (int j = 0; j < 16; ++j) a[j] = (f16)0.f;
    }
    int s     = s0 + (lane & 15);
    int bbase = k0 + ((lane < 16) ? 0 : 8);
    for (int j = 0; j < 16; ++j)
      b[j] = (f16)el[(size_t)(bbase + j + (j & 8)) * NSIMP + s];
    c = __builtin_amdgcn_wmma_f32_16x16x32_f16(false, a, false, b,
                                               (short)0, c, false, false);
  }
  if (lane < 16) {   // rows M=0..7 live in lanes 0..15, VGPRs 0..7
    int s = s0 + lane;
    for (int r = 0; r < NF; ++r)
      fs[((size_t)r * NL + l) * NSIMP + s] = c[r];
  }
}

// ---------------------------------------------------------------------------
// Branch: gather by mask, Conv1d(k=2,s=2), per-slice bitonic sort in LDS,
// exact level-2 signature (prefix telescopes: pref_t = path[t]-path[0]).
// One workgroup per (l, f).
// ---------------------------------------------------------------------------
__global__ __launch_bounds__(256) void k_branch(const float* __restrict__ fs,
                                                const int* __restrict__ mask,
                                                const float* __restrict__ cw,
                                                const float* __restrict__ cb,
                                                float* __restrict__ feat,
                                                int Mlen, int outOff) {
  const int Sd = 8;
  int l = blockIdx.x, f = blockIdx.y;
  int T = Mlen >> 1;
  __shared__ float dat[8][1024];
  __shared__ float w0[8], w1[8], bb[8];
  if (threadIdx.x < 8) {
    w0[threadIdx.x] = cw[(f * 8 + threadIdx.x) * 2 + 0];
    w1[threadIdx.x] = cw[(f * 8 + threadIdx.x) * 2 + 1];
    bb[threadIdx.x] = cb[f * 8 + threadIdx.x];
  }
  __syncthreads();
  const float* fsl = fs + ((size_t)f * NL + l) * NSIMP;
  const int*   ml  = mask + (size_t)l * Mlen;
  for (int t = threadIdx.x; t < T; t += 256) {
    float v0 = fsl[ml[2 * t]];
    float v1 = fsl[ml[2 * t + 1]];
    for (int s = 0; s < Sd; ++s) dat[s][t] = w0[s] * v0 + w1[s] * v1 + bb[s];
  }
  __syncthreads();
  // bitonic sort each slice row ascending (all 8 rows in parallel)
  for (int k = 2; k <= T; k <<= 1) {
    for (int j = k >> 1; j > 0; j >>= 1) {
      for (int idx = threadIdx.x; idx < Sd * T; idx += 256) {
        int s = idx / T, i = idx - s * T;
        int p = i ^ j;
        if (p > i) {
          float a = dat[s][i], b2 = dat[s][p];
          bool up = ((i & k) == 0);
          if ((a > b2) == up) { dat[s][i] = b2; dat[s][p] = a; }
        }
      }
      __syncthreads();
    }
  }
  // signature
  float* fout = feat + (size_t)l * 1152 + f * 144 + outOff;
  if (threadIdx.x < Sd * Sd) {
    int i = threadIdx.x >> 3, jc = threadIdx.x & 7;
    float acc = 0.f, acc2 = 0.f;
    float pi0 = dat[i][0];
    for (int t = 0; t < T - 1; ++t) {
      float dxi = dat[i][t + 1] - dat[i][t];
      float dxj = dat[jc][t + 1] - dat[jc][t];
      acc  += (dat[i][t] - pi0) * dxj;
      acc2 += dxi * dxj;
    }
    fout[Sd + i * Sd + jc] = acc + 0.5f * acc2;
  } else if (threadIdx.x < Sd * Sd + Sd) {
    int i = threadIdx.x - Sd * Sd;
    fout[i] = dat[i][T - 1] - dat[i][0];
  }
}

// ---------------------------------------------------------------------------
// Final: out = BN_L( feat @ final_w^T + final_b ).  Single workgroup.
// ---------------------------------------------------------------------------
__global__ __launch_bounds__(256) void k_final(const float* __restrict__ feat,
                                               const float* __restrict__ fw,
                                               const float* __restrict__ fb,
                                               const float* __restrict__ fg,
                                               const float* __restrict__ fbeta,
                                               float* __restrict__ out) {
  __shared__ float o[NL];
  int lane = threadIdx.x & 31, wid = threadIdx.x >> 5;
  for (int l = wid; l < NL; l += 8) {
    float s = 0.f;
    const float* xr = feat + (size_t)l * 1152;
    for (int k = lane; k < 1152; k += 32) s += xr[k] * fw[k];
    for (int off = 16; off > 0; off >>= 1) s += __shfl_xor(s, off, 32);
    if (lane == 0) o[l] = s + fb[0];
  }
  __syncthreads();
  __shared__ float mu_s, inv_s;
  if (threadIdx.x == 0) {
    float m = 0.f, q = 0.f;
    for (int l = 0; l < NL; ++l) m += o[l];
    m /= (float)NL;
    for (int l = 0; l < NL; ++l) { float d = o[l] - m; q += d * d; }
    q /= (float)NL;
    mu_s = m; inv_s = rsqrtf(q + EPSB);
  }
  __syncthreads();
  if (threadIdx.x < NL)
    out[threadIdx.x] = fg[0] * (o[threadIdx.x] - mu_s) * inv_s + fbeta[0];
}

// ---------------------------------------------------------------------------
extern "C" void kernel_launch(void* const* d_in, const int* in_sizes, int n_in,
                              void* d_out, int out_size, void* d_ws, size_t ws_size,
                              hipStream_t stream) {
  const float* e       = (const float*)d_in[0];
  const float* eig     = (const float*)d_in[1];
  const int*   mask0   = (const int*)  d_in[2];
  const int*   mask1   = (const int*)  d_in[3];
  const float* Win     = (const float*)d_in[4];
  const float* gin     = (const float*)d_in[6];
  const float* betain  = (const float*)d_in[7];
  const float* Wh      = (const float*)d_in[8];
  const float* bh      = (const float*)d_in[9];
  const float* gh      = (const float*)d_in[10];
  const float* betah   = (const float*)d_in[11];
  const float* Wout    = (const float*)d_in[12];
  const float* bout    = (const float*)d_in[13];
  const float* gout    = (const float*)d_in[14];
  const float* betaout = (const float*)d_in[15];
  const float* conv0_w = (const float*)d_in[16];
  const float* conv0_b = (const float*)d_in[17];
  const float* conv1_w = (const float*)d_in[18];
  const float* conv1_b = (const float*)d_in[19];
  const float* final_w = (const float*)d_in[20];
  const float* final_b = (const float*)d_in[21];
  const float* final_g = (const float*)d_in[22];
  const float* final_bt= (const float*)d_in[23];

  // workspace carve-up (all offsets 256B aligned)
  char* base = (char*)d_ws;
  f16*   ACT  = (f16*)  base; base += (size_t)NF * NTOT * HUNI * 2;   // 67 MB
  float* Y    = (float*)base; base += (size_t)NF * NTOT * HUNI * 4;   // 134 MB
  float* MU   = (float*)base; base += (size_t)NF * HUNI * 4;
  float* VAR  = (float*)base; base += (size_t)NF * HUNI * 4;
  float* EMU  = (float*)base; base += 256;
  float* EVAR = (float*)base; base += 256;
  float* OMU  = (float*)base; base += 256;
  float* OVAR = (float*)base; base += 256;
  float* YOUT = (float*)base; base += (size_t)NF * NTOT * 4;
  f16*   TRA  = (f16*)  base; base += (size_t)NL * NF * NEIGV * 2;
  float* FS   = (float*)base; base += (size_t)NF * NL * NSIMP * 4;    // 33.5 MB
  float* FEAT = (float*)base; base += (size_t)NL * 1152 * 4;

  // scalar stats of eigenvalue list, then analytic input-layer BN
  k_colstats<<<dim3(1, 1), 256, 0, stream>>>(e, EMU, EVAR, NTOT, 1);
  k_inlayer<<<4096, 256, 0, stream>>>(e, EMU, EVAR, Win, gin, betain, ACT);

  // 3 hidden layers: WMMA GEMM -> batch stats -> BN+ReLU (back into ACT)
  for (int layer = 0; layer < NHL; ++layer) {
    k_gemm512<<<dim3(NTOT / 64, HUNI / 32, NF), 256, 0, stream>>>(ACT, Wh, bh, Y, layer);
    k_colstats<<<dim3(HUNI, NF), 256, 0, stream>>>(Y, MU, VAR, NTOT, HUNI);
    k_bnrelu<<<4096, 256, 0, stream>>>(Y, MU, VAR, gh, betah, ACT, layer);
  }

  // output layer -> BN -> WMMA-ready tr fragments
  k_outdot<<<dim3(NTOT / 8, NF), 256, 0, stream>>>(ACT, Wout, bout, YOUT);
  k_colstats<<<dim3(1, NF), 256, 0, stream>>>(YOUT, OMU, OVAR, NTOT, 1);
  k_trbn<<<(NF * NTOT) / 256, 256, 0, stream>>>(YOUT, OMU, OVAR, gout, betaout, TRA);

  // fs = tr @ eigvec_sq  (single streaming pass over the 537 MB tensor)
  k_fs<<<dim3(NSIMP / 128, NL), 256, 0, stream>>>(TRA, eig, FS);

  // persistence branches: gather + conv + sort + level-2 signature
  k_branch<<<dim3(NL, NF), 256, 0, stream>>>(FS, mask0, conv0_w, conv0_b, FEAT, 2048, 0);
  k_branch<<<dim3(NL, NF), 256, 0, stream>>>(FS, mask1, conv1_w, conv1_b, FEAT, 1024, 72);

  // final linear + BN over L
  k_final<<<1, 256, 0, stream>>>(FEAT, final_w, final_b, final_g, final_bt,
                                 (float*)d_out);
}
